// VaeDecoderRobertaSelfAttention_21784074126117
// MI455X (gfx1250) — matrix operand order repaired and checked by reading, not compiled, not run on hardware
//
#include <hip/hip_runtime.h>

typedef __attribute__((ext_vector_type(16))) _Float16 v16h;
typedef __attribute__((ext_vector_type(8)))  _Float16 v8h;
typedef __attribute__((ext_vector_type(8)))  float    v8f;
typedef __attribute__((ext_vector_type(4)))  unsigned int u32x4;
typedef __attribute__((ext_vector_type(8)))  int      i32x8;
typedef __attribute__((ext_vector_type(4)))  int      i32x4;

#define BB 4
#define SS 2048
#define DD 1024
#define HH 16
#define HDIM 64
#define TT (SS + 1)    // 2049 keys: t=0..2047 seq, t=2048 latent
#define TP 2080        // padded key axis for transposed V

static __device__ __forceinline__ v8f wmma_f16(v16h a, v16h b, v8f c) {
    return __builtin_amdgcn_wmma_f32_16x16x32_f16(false, a, false, b, (short)0, c, false, false);
}

static __device__ __forceinline__ v16h cat8(v8h lo, v8h hi) {
    return __builtin_shufflevector(lo, hi, 0, 1, 2, 3, 4, 5, 6, 7,
                                           8, 9, 10, 11, 12, 13, 14, 15);
}

// Generic -> LDS(addrspace 3) byte offset for TDM descriptors.
static __device__ __forceinline__ unsigned lds_off_of(const void* p) {
    return (unsigned)(uintptr_t)(const __attribute__((address_space(3))) void*)p;
}

// 2-D TDM tile load: global (row-major, stride0 in 8B units) -> LDS contiguous.
// D# group0: count=1 | lds_addr | global_addr | type=2
// D# group1: data_size=3 (8B) | tensor_dim0/1 | tile_dim0/1 | tensor_dim0_stride
// This toolchain exposes the 6-arg builtin: (u32x4, i32x8, i32x4, i32x4, i32x8, i32 cpol).
static __device__ __forceinline__ void tdm_load_2d(
        unsigned lds_off, const void* g, unsigned tile0_units, unsigned tile1,
        unsigned tdim0_units, unsigned tdim1, unsigned stride0_units) {
    unsigned long long ga = (unsigned long long)(uintptr_t)g;
    u32x4 g0;
    g0[0] = 1u;                                             // count=1, user mode
    g0[1] = lds_off;                                        // lds_addr (bytes)
    g0[2] = (unsigned)ga;                                   // global_addr[31:0]
    g0[3] = (unsigned)((ga >> 32) & 0x01ffffffu) | (2u << 30); // addr[56:32], type=2
    i32x8 g1;
    g1[0] = (int)(3u << 16);                                // data_size = 8B
    g1[1] = (int)((tdim0_units & 0xffffu) << 16);           // tensor_dim0[15:0]
    g1[2] = (int)(((tdim0_units >> 16) & 0xffffu) |
                  ((tdim1 & 0xffffu) << 16));               // tdim0[31:16]|tdim1[15:0]
    g1[3] = (int)(((tdim1 >> 16) & 0xffffu) |
                  ((tile0_units & 0xffffu) << 16));         // tdim1[31:16]|tile_dim0
    g1[4] = (int)(tile1 & 0xffffu);                         // tile_dim1 (tile_dim2=0)
    g1[5] = (int)stride0_units;                             // dim0 stride[31:0]
    g1[6] = 0;                                              // stride[47:32]=0, dim1_stride=0
    g1[7] = 0;
    i32x4 z4 = (i32x4){0, 0, 0, 0};                         // groups 2/3 unused (2-D tile)
    i32x8 z8 = (i32x8){0, 0, 0, 0, 0, 0, 0, 0};
    __builtin_amdgcn_tensor_load_to_lds(g0, g1, z4, z4, z8, 0);
}

// ---------------------------------------------------------------------------
// Latent -> key slot 2048 of Kh (row-major) and Vt (transposed), as f16.
// ---------------------------------------------------------------------------
__global__ void latent_copy_kernel(const float* __restrict__ latent,
                                   _Float16* __restrict__ Kh, _Float16* __restrict__ Vt) {
    int i = blockIdx.x * 256 + threadIdx.x;
    if (i < BB * DD) {
        int b = i / DD, d = i % DD;
        _Float16 v = (_Float16)latent[i];
        Kh[((size_t)b * TT + SS) * DD + d] = v;
        Vt[((size_t)b * DD + d) * TP + SS] = v;
    }
}

// ---------------------------------------------------------------------------
// QKV projection: Y = X @ W^T + bias -> f16 workspace.  grid.z: 0=Q,1=K,2=V.
// Q pre-scaled by 1/sqrt(HD).  V stored transposed [B, D, TP].
// ---------------------------------------------------------------------------
__global__ __launch_bounds__(256) void qkv_proj_kernel(
        const float* __restrict__ X,
        const float* __restrict__ Wq, const float* __restrict__ bq,
        const float* __restrict__ Wk, const float* __restrict__ bk,
        const float* __restrict__ Wv, const float* __restrict__ bv,
        _Float16* __restrict__ Qh, _Float16* __restrict__ Kh,
        _Float16* __restrict__ Vt) {
    const int z = blockIdx.z;
    const float* W    = (z == 0) ? Wq : (z == 1) ? Wk : Wv;
    const float* bias = (z == 0) ? bq : (z == 1) ? bk : bv;

    const int wave = threadIdx.x >> 5;
    const int l    = threadIdx.x & 31;
    const int half = l >> 4;
    const int n16  = l & 15;

    const int row0 = blockIdx.x * 128 + wave * 16;
    const int col0 = blockIdx.y * 64;

    v8f acc[4];
#pragma unroll
    for (int dt = 0; dt < 4; ++dt)
        acc[dt] = (v8f){0.f, 0.f, 0.f, 0.f, 0.f, 0.f, 0.f, 0.f};

    for (int k0 = 0; k0 < DD; k0 += 32) {
        const float* xr = X + (size_t)(row0 + n16) * DD + k0 + half * 16;
        v16h a;
#pragma unroll
        for (int j = 0; j < 4; ++j) {
            float2 f0 = *(const float2*)(xr + 2 * j);
            float2 f1 = *(const float2*)(xr + 8 + 2 * j);
            a[2 * j]     = (_Float16)f0.x; a[2 * j + 1]     = (_Float16)f0.y;
            a[8 + 2 * j] = (_Float16)f1.x; a[8 + 2 * j + 1] = (_Float16)f1.y;
        }
#pragma unroll
        for (int dt = 0; dt < 4; ++dt) {
            const float* wr = W + (size_t)(col0 + dt * 16 + n16) * DD + k0 + half * 16;
            v16h bfr;
#pragma unroll
            for (int j = 0; j < 8; ++j) {
                float2 f = *(const float2*)(wr + 2 * j);
                bfr[2 * j] = (_Float16)f.x; bfr[2 * j + 1] = (_Float16)f.y;
            }
            acc[dt] = wmma_f16(a, bfr, acc[dt]);
        }
    }

    const int bidx = row0 >> 11;
    const int s0   = (row0 & 2047) + half * 8;

#pragma unroll
    for (int dt = 0; dt < 4; ++dt) {
        const int col = col0 + dt * 16 + n16;
        const float bsc = bias[col];
        if (z == 0) {                       // Q: [B*S, D], scaled by 0.125
#pragma unroll
            for (int j = 0; j < 8; ++j) {
                const int r = row0 + half * 8 + j;
                Qh[(size_t)r * DD + col] = (_Float16)((acc[dt][j] + bsc) * 0.125f);
            }
        } else if (z == 1) {                // K: [B, T, D], seq rows t = s
#pragma unroll
            for (int j = 0; j < 8; ++j)
                Kh[((size_t)bidx * TT + s0 + j) * DD + col] =
                    (_Float16)(acc[dt][j] + bsc);
        } else {                            // V: transposed [B, D, TP], one b128/dt
            v8h pk;
#pragma unroll
            for (int j = 0; j < 8; ++j) pk[j] = (_Float16)(acc[dt][j] + bsc);
            *(v8h*)(Vt + ((size_t)bidx * DD + col) * TP + s0) = pk;
        }
    }
}

// ---------------------------------------------------------------------------
// Flash attention, TDM-staged K/V with double buffering.
// One wave per (b, h, 16-query tile); key blocks of 32.
// ---------------------------------------------------------------------------
__global__ __launch_bounds__(128) void attn_kernel(
        const _Float16* __restrict__ Q, const _Float16* __restrict__ Km,
        const _Float16* __restrict__ Vt, const float* __restrict__ mask,
        float* __restrict__ out) {
    __shared__ _Float16 kbuf[4][2][32 * 64];   // 32 keys x 64 dims per buffer
    __shared__ _Float16 vbuf[4][2][64 * 32];   // 64 dims x 32 keys per buffer
    __shared__ _Float16 pbuf[4][16 * 32];

    const int wave = threadIdx.x >> 5;
    const int l    = threadIdx.x & 31;
    const int half = l >> 4;
    const int n16  = l & 15;

    const int qt = blockIdx.x * 4 + wave;
    const int bh = qt / (SS / 16);
    const int qi = qt % (SS / 16);
    const int b  = bh / HH;
    const int h  = bh % HH;
    const int q0 = qi * 16;
    const size_t hoff = (size_t)h * HDIM;

    const _Float16* Qb = Q  + (size_t)b * SS * DD;
    const _Float16* Kb = Km + (size_t)b * TT * DD + hoff;      // [t][64] slice
    const _Float16* Vb = Vt + ((size_t)b * DD + hoff) * TP;    // [64][TP] slice
    const float*    Mb = mask + (size_t)b * SS * SS;

    const unsigned koff[2] = { lds_off_of(&kbuf[wave][0][0]), lds_off_of(&kbuf[wave][1][0]) };
    const unsigned voff[2] = { lds_off_of(&vbuf[wave][0][0]), lds_off_of(&vbuf[wave][1][0]) };

    // Q fragments (pre-scaled at projection time)
    v16h aq[2];
    {
        const _Float16* qrow = Qb + (size_t)(q0 + n16) * DD + hoff + half * 16;
#pragma unroll
        for (int g = 0; g < 2; ++g)
            aq[g] = cat8(*(const v8h*)(qrow + g * 32),
                         *(const v8h*)(qrow + g * 32 + 8));
    }

    float rowmax[8], rowsum[8];
    v8f acc[4];
#pragma unroll
    for (int j = 0; j < 8; ++j) { rowmax[j] = -1e30f; rowsum[j] = 0.f; }
#pragma unroll
    for (int dt = 0; dt < 4; ++dt)
        acc[dt] = (v8f){0.f, 0.f, 0.f, 0.f, 0.f, 0.f, 0.f, 0.f};

    _Float16* pl = &pbuf[wave][0];

    // Prologue: stage key-block 0 into buffer 0.
    // K tile: 32 rows x 128B (16 x 8B units), row stride D*2/8 = 256 units.
    // V tile: 64 rows x  64B ( 8 x 8B units), row stride TP*2/8 = 520 units.
    tdm_load_2d(koff[0], Kb, 16, 32, 16, (unsigned)TT, (DD * 2) / 8);
    tdm_load_2d(voff[0], Vb,  8, 64,  8, 64u,          (TP * 2) / 8);

    int buf = 0;
    for (int t0 = 0; t0 < TT; t0 += 32) {
        // Prefetch next key-block into the other buffer, then wait for current.
        if (t0 + 32 < TT) {
            tdm_load_2d(koff[buf ^ 1], Kb + (size_t)(t0 + 32) * DD,
                        16, 32, 16, (unsigned)(TT - t0 - 32), (DD * 2) / 8);
            tdm_load_2d(voff[buf ^ 1], Vb + (t0 + 32),
                        8, 64, 8, 64u, (TP * 2) / 8);
            __builtin_amdgcn_s_wait_tensorcnt(2);   // oldest pair (current) done
        } else {
            __builtin_amdgcn_s_wait_tensorcnt(0);
        }
        asm volatile("" ::: "memory");

        const _Float16* kt_ = &kbuf[wave][buf][0];
        const _Float16* vt_ = &vbuf[wave][buf][0];

        // ---- scores: two 16x16 tiles, contraction over HD=64 (2 WMMA each) ----
        v8f s[2];
#pragma unroll
        for (int kt = 0; kt < 2; ++kt) {
            const _Float16* krow = kt_ + (kt * 16 + n16) * 64 + half * 16;
            v8f c = (v8f){0.f, 0.f, 0.f, 0.f, 0.f, 0.f, 0.f, 0.f};
#pragma unroll
            for (int g = 0; g < 2; ++g)
                c = wmma_f16(aq[g], *(const v16h*)(krow + g * 32), c);
            s[kt] = c;
        }

        // ---- additive mask: seq keys -> mask[q][t]; latent -> 0; tail -> -inf ----
#pragma unroll
        for (int kt = 0; kt < 2; ++kt) {
            const int t = t0 + kt * 16 + n16;
            if (t > SS) {
#pragma unroll
                for (int j = 0; j < 8; ++j) s[kt][j] = -1e30f;
            } else if (t < SS) {
                const float* mcol = Mb + t;
#pragma unroll
                for (int j = 0; j < 8; ++j)
                    s[kt][j] += mcol[(size_t)(q0 + half * 8 + j) * SS];
            } // t == SS (latent): +0
        }

        // ---- online softmax (row lives across the 16 lanes of this half) ----
#pragma unroll
        for (int j = 0; j < 8; ++j) {
            float mx = fmaxf(s[0][j], s[1][j]);
#pragma unroll
            for (int d = 1; d < 16; d <<= 1) mx = fmaxf(mx, __shfl_xor(mx, d, 16));
            const float mnew = fmaxf(rowmax[j], mx);
            const float corr = __expf(rowmax[j] - mnew);
            rowmax[j] = mnew;
            const float p0 = __expf(s[0][j] - mnew);
            const float p1 = __expf(s[1][j] - mnew);
            s[0][j] = p0; s[1][j] = p1;
            float ps = p0 + p1;
#pragma unroll
            for (int d = 1; d < 16; d <<= 1) ps += __shfl_xor(ps, d, 16);
            rowsum[j] = rowsum[j] * corr + ps;
#pragma unroll
            for (int dt = 0; dt < 4; ++dt) acc[dt][j] *= corr;
        }

        // ---- P: C-layout -> LDS f16 (16x32 row-major) -> A-layout ----
#pragma unroll
        for (int kt = 0; kt < 2; ++kt)
#pragma unroll
            for (int j = 0; j < 8; ++j)
                pl[(half * 8 + j) * 32 + kt * 16 + n16] = (_Float16)s[kt][j];

        const _Float16* pr = pl + n16 * 32 + half * 16;
        v16h ap = cat8(*(const v8h*)(pr), *(const v8h*)(pr + 8));

        // ---- P @ V from staged LDS tile: one v16h ds load per d-tile ----
#pragma unroll
        for (int dt = 0; dt < 4; ++dt) {
            const _Float16* vrow = vt_ + (dt * 16 + n16) * 32 + half * 16;
            acc[dt] = wmma_f16(ap, *(const v16h*)vrow, acc[dt]);
        }

        buf ^= 1;
    }

    // ---- normalize + store: out[b][q0+m][h*64 + dt*16 + n16] (f32) ----
#pragma unroll
    for (int j = 0; j < 8; ++j) {
        const float inv = 1.f / rowsum[j];
        const int m = half * 8 + j;
        float* orow = out + ((size_t)b * SS + q0 + m) * DD + hoff;
#pragma unroll
        for (int dt = 0; dt < 4; ++dt)
            orow[dt * 16 + n16] = acc[dt][j] * inv;
    }
}

// ---------------------------------------------------------------------------
extern "C" void kernel_launch(void* const* d_in, const int* in_sizes, int n_in,
                              void* d_out, int out_size, void* d_ws, size_t ws_size,
                              hipStream_t stream) {
    const float* hidden = (const float*)d_in[0];
    const float* mask   = (const float*)d_in[1];
    const float* latent = (const float*)d_in[2];
    const float* Wq = (const float*)d_in[3];
    const float* bq = (const float*)d_in[4];
    const float* Wk = (const float*)d_in[5];
    const float* bk = (const float*)d_in[6];
    const float* Wv = (const float*)d_in[7];
    const float* bv = (const float*)d_in[8];
    float* out = (float*)d_out;

    _Float16* Qh = (_Float16*)d_ws;                  // [B*S, D]
    _Float16* Kh = Qh + (size_t)BB * SS * DD;        // [B, T, D]
    _Float16* Vt = Kh + (size_t)BB * TT * DD;        // [B, D, TP]

    latent_copy_kernel<<<(BB * DD + 255) / 256, 256, 0, stream>>>(latent, Kh, Vt);

    dim3 pgrid(BB * SS / 128, DD / 64, 3);
    qkv_proj_kernel<<<pgrid, 256, 0, stream>>>(hidden, Wq, bq, Wk, bk, Wv, bv,
                                               Qh, Kh, Vt);

    const int qtiles = BB * HH * (SS / 16);          // 8192
    attn_kernel<<<qtiles / 4, 128, 0, stream>>>(Qh, Kh, Vt, mask, out);
}
